// ContrastiveLoss_70007966924843
// MI455X (gfx1250) — compile-verified, matrix operand order checked by reading, not run on hardware
//
#include <hip/hip_runtime.h>
#include <hip/hip_bf16.h>

// NT-Xent (SimCLR) contrastive loss, fused for gfx1250.
// zh = (sqrt(2)/max(||z||,eps)) * z  stored f16  =>  zh_i . zh_j == logits[i][j]
// loss = mean_i( log( sum_{j!=i} exp(l_ij) ) - l_{i,target(i)} )  (|l| <= 2, no max pass)

typedef _Float16 v16h __attribute__((ext_vector_type(16)));
typedef _Float16 v8h  __attribute__((ext_vector_type(8)));
typedef float    v8f  __attribute__((ext_vector_type(8)));

#define DIM    256
#define KTILES 8            // 256 / 32
#define CHUNK  64           // columns (zh rows) staged per LDS chunk
#define PAD    8            // halves of pad per LDS row (stride 132 dwords -> conflict-free)
#define LROW   (DIM + PAD)  // 264 halves = 528 B per row

// 16-row x 32-K f16 WMMA fragment from global (A side of the Gram matrix).
__device__ __forceinline__ v16h gfrag(const _Float16* __restrict__ zh,
                                      int base_row, int k0, int lane) {
    int row = base_row + (lane & 15);
    int col = k0 + ((lane >> 4) << 4);
    return *reinterpret_cast<const v16h*>(zh + (size_t)row * DIM + col);
}

// Same fragment shape out of the padded LDS chunk (B side).
__device__ __forceinline__ v16h lfrag(const _Float16* buf, int c_local, int k0, int lane) {
    int row = c_local + (lane & 15);
    int col = k0 + ((lane >> 4) << 4);
    return *reinterpret_cast<const v16h*>(buf + row * LROW + col);
}

// Async DMA one 64-row chunk (this wave's 8 rows) global -> LDS, no VGPR staging.
// One B128 per lane covers a 512 B row per instruction; tracked by ASYNCcnt.
__device__ __forceinline__ void async_stage_chunk(const _Float16* __restrict__ zh,
                                                  int c0, unsigned lds_base,
                                                  int wave, int lane) {
    #pragma unroll
    for (int j = 0; j < 8; ++j) {
        int row = wave * 8 + j;
        const _Float16* gp = zh + (size_t)(c0 + row) * DIM + lane * 8;
        unsigned lp = lds_base + (unsigned)(row * (LROW * 2)) + lane * 16;
        asm volatile("global_load_async_to_lds_b128 %0, %1, off"
                     :: "v"(lp), "v"(gp) : "memory");
    }
}

// ---------------------------------------------------------------------------
// Kernel 1: row-normalize concat(z_i, z_j), fold sqrt(1/T)=sqrt(2), emit f16.
// ---------------------------------------------------------------------------
__global__ void nrm_kernel(const float* __restrict__ zi,
                           const float* __restrict__ zj,
                           _Float16* __restrict__ zh, int Brows) {
    int wave = (blockIdx.x * blockDim.x + threadIdx.x) >> 5;
    int lane = threadIdx.x & 31;
    int N2 = 2 * Brows;
    if (wave >= N2) return;
    const float* src = (wave < Brows) ? (zi + (size_t)wave * DIM)
                                      : (zj + (size_t)(wave - Brows) * DIM);
    const float4* src4 = reinterpret_cast<const float4*>(src);
    float4 a = src4[lane * 2 + 0];
    float4 b = src4[lane * 2 + 1];
    float ss = a.x * a.x + a.y * a.y + a.z * a.z + a.w * a.w
             + b.x * b.x + b.y * b.y + b.z * b.z + b.w * b.w;
    #pragma unroll
    for (int m = 16; m >= 1; m >>= 1) ss += __shfl_xor(ss, m, 32);
    float s = 1.41421356237309515f / fmaxf(sqrtf(ss), 1e-8f);
    v8h o;
    o[0] = (_Float16)(a.x * s); o[1] = (_Float16)(a.y * s);
    o[2] = (_Float16)(a.z * s); o[3] = (_Float16)(a.w * s);
    o[4] = (_Float16)(b.x * s); o[5] = (_Float16)(b.y * s);
    o[6] = (_Float16)(b.z * s); o[7] = (_Float16)(b.w * s);
    *reinterpret_cast<v8h*>(zh + (size_t)wave * DIM + lane * 8) = o;
}

// ---------------------------------------------------------------------------
// Kernel 2: fused Gram + exp + masked row-sum.
// Block = 8 waves, each owns 16 rows (128 rows/block). 64-column chunks are
// async-DMA'd into double-buffered LDS (global_load_async_to_lds_b128, no
// VGPR staging -> no spills); the copy of chunk k+1 runs under chunk k's
// 32 WMMAs/wave, drained with s_wait_asynccnt before the chunk barrier.
// ---------------------------------------------------------------------------
__global__ void __launch_bounds__(256)
gram_kernel(const _Float16* __restrict__ zh,
            float* __restrict__ rowS,
            int N2, int colsPerGroup) {
    __shared__ _Float16 lds_buf[2][CHUNK * LROW];   // 2 x 33 KB

    int tid  = threadIdx.x;
    int lane = tid & 31;
    int wave = tid >> 5;                         // 0..7
    int r0 = (blockIdx.x * 8 + wave) * 16;       // this wave's 16-row tile
    int c_begin = blockIdx.y * colsPerGroup;

    // LDS byte offsets of the two staging buffers (flat addr low 32 bits).
    unsigned lbase0 = (unsigned)(unsigned long long)&lds_buf[0][0];
    unsigned lbase1 = (unsigned)(unsigned long long)&lds_buf[1][0];

    // A fragments for all 8 K-slices (held for the whole kernel).
    v16h a[KTILES];
    #pragma unroll
    for (int k = 0; k < KTILES; ++k) a[k] = gfrag(zh, r0, k * 32, lane);

    float accS[8];
    #pragma unroll
    for (int v = 0; v < 8; ++v) accS[v] = 0.0f;

    // C/D layout: VGPR v, lanes 0-15 -> M=v, N=lane; lanes 16-31 -> M=v+8, N=lane-16
    int m_base = r0 + ((lane >> 4) << 3);
    int n_lane = lane & 15;

    // ---- prologue: DMA chunk 0 into buffer 0 ----
    async_stage_chunk(zh, c_begin, lbase0, wave, lane);
    asm volatile("s_wait_asynccnt 0x0" ::: "memory");
    __syncthreads();

    int nChunks = colsPerGroup / CHUNK;
    for (int ch = 0; ch < nChunks; ++ch) {
        int c0 = c_begin + ch * CHUNK;
        const _Float16* buf = &lds_buf[ch & 1][0];
        bool more = (ch + 1) < nChunks;

        // Kick off the next chunk's DMA; it completes under the WMMAs below.
        if (more)
            async_stage_chunk(zh, c0 + CHUNK, (ch & 1) ? lbase0 : lbase1, wave, lane);

        // 4 column tiles as 2 independent accumulation chains, K software-pipelined.
        #pragma unroll
        for (int ct = 0; ct < CHUNK / 16; ct += 2) {
            v8f acc0 = {}, acc1 = {};
            v16h b0 = lfrag(buf, ct * 16,      0, lane);
            v16h b1 = lfrag(buf, ct * 16 + 16, 0, lane);
            #pragma unroll
            for (int k = 0; k < KTILES; ++k) {
                v16h nb0, nb1;
                if (k + 1 < KTILES) {
                    nb0 = lfrag(buf, ct * 16,      (k + 1) * 32, lane);
                    nb1 = lfrag(buf, ct * 16 + 16, (k + 1) * 32, lane);
                }
                acc0 = __builtin_amdgcn_wmma_f32_16x16x32_f16(
                           false, a[k], false, b0, (short)0, acc0, false, false);
                acc1 = __builtin_amdgcn_wmma_f32_16x16x32_f16(
                           false, a[k], false, b1, (short)0, acc1, false, false);
                b0 = nb0; b1 = nb1;
            }
            int n0 = c0 + ct * 16 + n_lane;
            #pragma unroll
            for (int v = 0; v < 8; ++v) {
                float e0 = __expf(acc0[v]);           // |logit| <= 2
                float e1 = __expf(acc1[v]);
                if (m_base + v == n0)      e0 = 0.0f; // mask diagonal
                if (m_base + v == n0 + 16) e1 = 0.0f;
                accS[v] += e0 + e1;
            }
        }

        // Ensure next chunk's DMA landed in LDS before anyone reads it.
        if (more)
            asm volatile("s_wait_asynccnt 0x0" ::: "memory");
        __syncthreads();
    }

    // Reduce across the 16 lanes of each half (N dimension), store row sums.
    #pragma unroll
    for (int v = 0; v < 8; ++v) {
        float s = accS[v];
        s += __shfl_xor(s, 1, 16);
        s += __shfl_xor(s, 2, 16);
        s += __shfl_xor(s, 4, 16);
        s += __shfl_xor(s, 8, 16);
        if (n_lane == 0) rowS[(size_t)blockIdx.y * N2 + m_base + v] = s;
    }
}

// ---------------------------------------------------------------------------
// Kernel 3: loss = mean_i( log(sum_g rowS[g][i]) - dot(zh_i, zh_partner) ).
// ---------------------------------------------------------------------------
__global__ void finalize_kernel(const _Float16* __restrict__ zh,
                                const float* __restrict__ rowS,
                                float* __restrict__ out,
                                int Brows, int nGroups) {
    int lane = threadIdx.x & 31;
    int wave = threadIdx.x >> 5;     // 0..7
    int N2 = 2 * Brows;
    __shared__ float partial[8];
    float local = 0.0f;
    for (int r = wave; r < N2; r += 8) {
        int p = (r < Brows) ? (r + Brows) : (r - Brows);
        v8h va = *reinterpret_cast<const v8h*>(zh + (size_t)r * DIM + lane * 8);
        v8h vb = *reinterpret_cast<const v8h*>(zh + (size_t)p * DIM + lane * 8);
        float dot = 0.0f;
        #pragma unroll
        for (int t = 0; t < 8; ++t) dot += (float)va[t] * (float)vb[t];
        #pragma unroll
        for (int m = 16; m >= 1; m >>= 1) dot += __shfl_xor(dot, m, 32);
        if (lane == 0) {
            float S = 0.0f;
            for (int g = 0; g < nGroups; ++g) S += rowS[(size_t)g * N2 + r];
            local += __logf(S) - dot;     // dot already includes the 1/T folding
        }
    }
    if (lane == 0) partial[wave] = local;
    __syncthreads();
    if (threadIdx.x == 0) {
        float tot = 0.0f;
        #pragma unroll
        for (int w = 0; w < 8; ++w) tot += partial[w];
        out[0] = tot / (float)N2;
    }
}

extern "C" void kernel_launch(void* const* d_in, const int* in_sizes, int n_in,
                              void* d_out, int out_size, void* d_ws, size_t ws_size,
                              hipStream_t stream) {
    const float* zi = (const float*)d_in[0];
    const float* zj = (const float*)d_in[1];
    int Brows = in_sizes[0] / DIM;   // 4096
    int N2 = 2 * Brows;              // 8192

    _Float16* zh   = (_Float16*)d_ws;                               // 4 MB
    float*    rowS = (float*)((char*)d_ws + (size_t)N2 * DIM * sizeof(_Float16));

    const int nGroups = 4;
    int colsPerGroup = N2 / nGroups;                                // 2048

    dim3 nblocks((N2 + 7) / 8);
    nrm_kernel<<<nblocks, 256, 0, stream>>>(zi, zj, zh, Brows);

    dim3 gblocks(N2 / 128, nGroups);                                // (64, 4)
    gram_kernel<<<gblocks, 256, 0, stream>>>(zh, rowS, N2, colsPerGroup);

    finalize_kernel<<<1, 256, 0, stream>>>(zh, rowS, (float*)d_out, Brows, nGroups);
}
